// CausalSelfAttention_5111011082658
// MI455X (gfx1250) — matrix operand order repaired and checked by reading, not compiled, not run on hardware
//
#include <hip/hip_runtime.h>

typedef __attribute__((ext_vector_type(16))) __bf16 v16bf;
typedef __attribute__((ext_vector_type(8)))  float  v8f;
typedef __attribute__((ext_vector_type(4)))  unsigned int v4u;
typedef __attribute__((ext_vector_type(8)))  int  v8i;
typedef __attribute__((ext_vector_type(4)))  int  v4i;

#define TT 2048
#define NH 16
#define DH 64
#define CC 1024

#if defined(__has_builtin)
#if __has_builtin(__builtin_amdgcn_tensor_load_to_lds) && __has_builtin(__builtin_amdgcn_s_wait_tensorcnt)
#define HAVE_TDM 1
#endif
#endif
#ifndef HAVE_TDM
#define HAVE_TDM 0
#endif

__device__ __forceinline__ unsigned short f32_to_bf16(float f) {
  union { float f; unsigned int u; } c; c.f = f;
  unsigned int u = c.u;
  u += 0x7FFFu + ((u >> 16) & 1u);   // round-to-nearest-even
  return (unsigned short)(u >> 16);
}

union FragAB { v16bf v; unsigned int u[8]; };

// A-fragment (16xK=32, bf16): lane m = lane&15, half g = lane>>4.
// VGPR v (pairs in dword): k = (v%4)*2 + g*8 + (v/4)*16  (ISA 7.12.2)
__device__ __forceinline__ v16bf load_afrag(const unsigned short* base, int strideE,
                                            int mOff, int kOff) {
  int lane = threadIdx.x & 31;
  int m = mOff + (lane & 15);
  int g = lane >> 4;
  const unsigned int* p = (const unsigned int*)(base + (size_t)m * strideE + kOff);
  FragAB r;
#pragma unroll
  for (int i = 0; i < 4; ++i) {
    r.u[i]     = p[g * 4 + i];
    r.u[4 + i] = p[8 + g * 4 + i];
  }
  return r.v;
}

// B-fragment (K=32 x 16) read from an n-major ("B-transposed") LDS tile:
// lane n = lane&15, k = g*8 + {0..7} and 16 + g*8 + {0..7} (same order as A).
__device__ __forceinline__ v16bf load_bfrag(const unsigned short* baseT, int strideE,
                                            int nOff, int kOff) {
  int lane = threadIdx.x & 31;
  int n = nOff + (lane & 15);
  int g = lane >> 4;
  const unsigned int* p = (const unsigned int*)(baseT + (size_t)n * strideE + kOff);
  FragAB r;
#pragma unroll
  for (int i = 0; i < 4; ++i) {
    r.u[i]     = p[g * 4 + i];
    r.u[4 + i] = p[8 + g * 4 + i];
  }
  return r.v;
}

__device__ __forceinline__ v8f wmma_bf16(v16bf a, v16bf b, v8f c) {
  return __builtin_amdgcn_wmma_f32_16x16x32_bf16(false, a, false, b,
                                                 (short)0, c, false, false);
}

#if HAVE_TDM
// ---------------------------------------------------------------------------
// Tensor Data Mover: 2-D tile load (global -> LDS) with per-row LDS padding.
// D# packing per CDNA5 ISA 8.3/8.4:
//   group0: count=1 | lds_addr | global_addr[56:0] | type=2
//   group1: w0 = {data_size=1(2B), pad_enable, pad_interval, pad_amount}
//           tensor_dim0/1, tile_dim0/1, tensor_dim0_stride (elements)
//   groups 2/3 (+trailing words): zero (2-D tensor)
// flags_w0 precomputed by caller.
// ---------------------------------------------------------------------------
__device__ __forceinline__ void tdm_load_2d(unsigned lds_off,
                                            const unsigned short* gptr,
                                            unsigned tensor_d0, unsigned tensor_d1,
                                            unsigned tile_d0, unsigned tile_d1,
                                            unsigned stride0_elems,
                                            unsigned flags_w0) {
  unsigned long long ga = (unsigned long long)gptr;
  v4u g0 = { 1u, lds_off, (unsigned)ga,
             (unsigned)((ga >> 32) & 0x01FFFFFFu) | 0x80000000u };
  v8i g1 = { (int)flags_w0,
             (int)((tensor_d0 & 0xFFFFu) << 16),
             (int)((tensor_d0 >> 16) | ((tensor_d1 & 0xFFFFu) << 16)),
             (int)((tensor_d1 >> 16) | ((tile_d0 & 0xFFFFu) << 16)),
             (int)(tile_d1 & 0xFFFFu),
             (int)stride0_elems,
             0, 0 };
  v4i gz4 = { 0, 0, 0, 0 };
  v8i gz8 = { 0, 0, 0, 0, 0, 0, 0, 0 };
  __builtin_amdgcn_tensor_load_to_lds(g0, g1, gz4, gz4, gz8, 0);
}

__device__ __forceinline__ unsigned lds_off_of(const void* p) {
  // Generic pointers to LDS carry the LDS byte offset in the low 32 bits.
  return (unsigned)(unsigned long long)p;
}

// data_size=2B (code 1) << 16 | pad_enable << 20 | pad_interval << 22 | pad_amount << 25
#define TDM_FLAGS_ROW32  0x00D10000u  // pad after 16 dwords (row=32 bf16), +1 dword -> stride 34
#define TDM_FLAGS_ROW64  0x01110000u  // pad after 32 dwords (row=64 bf16), +1 dword -> stride 66
#endif

__global__ void cvt_f32_bf16_kernel(const float* __restrict__ src,
                                    unsigned short* __restrict__ dst, int n) {
  for (int i = blockIdx.x * blockDim.x + threadIdx.x; i < n;
       i += gridDim.x * blockDim.x)
    dst[i] = f32_to_bf16(src[i]);
}

// ---------------------------------------------------------------------------
// Tiled bf16 WMMA GEMM: C[M,N] = A[M,K] * B[K,N] + bias.
// Block tile 128x64, 8 waves (4 along M, 2 along N), k-step 32.
// A tile (no transpose) fetched by the Tensor Data Mover; B tile is staged
// transposed by the waves (TDM cannot transpose).
// EPI==0: scatter to Q/K/V head-major bf16.  EPI==1: fp32 row-major out.
// ---------------------------------------------------------------------------
template <int EPI>
__global__ __launch_bounds__(256) void gemm_bf16_kernel(
    const unsigned short* __restrict__ A,
    const unsigned short* __restrict__ Bw,
    const float* __restrict__ bias,
    int N, int K,
    unsigned short* __restrict__ outQ,
    unsigned short* __restrict__ outK,
    unsigned short* __restrict__ outV,
    float* __restrict__ outF) {
  __shared__ unsigned short sA[128 * 34];  // row-major, padded stride (TDM pad)
  __shared__ unsigned short sB[64 * 34];   // n-major (transposed), padded stride

  const int tid  = threadIdx.x;
  const int lane = tid & 31, wid = tid >> 5;
  const int g = lane >> 4, n16 = lane & 15;
  const int mBase = blockIdx.y * 128, nBase = blockIdx.x * 64;
  const int wm = (wid & 3) * 32, wn = (wid >> 2) * 32;

  v8f zero = {};
  v8f acc[2][2];
  acc[0][0] = zero; acc[0][1] = zero; acc[1][0] = zero; acc[1][1] = zero;

  const int kTiles = K >> 5;
  for (int kb = 0; kb < kTiles; ++kb) {
    __syncthreads();
#if HAVE_TDM
    // A tile 128x32 via TDM (one DMA per block, issued by wave 0)
    if (wid == 0) {
      tdm_load_2d(lds_off_of(sA), A + (size_t)mBase * K + kb * 32,
                  /*tensor_d0=*/32, /*tensor_d1=*/128,
                  /*tile_d0=*/32, /*tile_d1=*/128,
                  /*stride0=*/(unsigned)K, TDM_FLAGS_ROW32);
    }
#else
#pragma unroll
    for (int s = tid; s < 512; s += 256) {
      int row = s >> 2, c8 = s & 3;
      const uint4 d = *(const uint4*)(A + (size_t)(mBase + row) * K + kb * 32 + c8 * 8);
      unsigned int* dst = (unsigned int*)(sA + row * 34 + c8 * 8);
      dst[0] = d.x; dst[1] = d.y; dst[2] = d.z; dst[3] = d.w;
    }
#endif
    // B tile 32x64 -> store transposed (n-major), overlaps the TDM transfer
    {
      int row = tid >> 3, grp = tid & 7;
      union { uint4 q; unsigned short e[8]; } u;
      u.q = *(const uint4*)(Bw + (size_t)(kb * 32 + row) * N + nBase + grp * 8);
#pragma unroll
      for (int i = 0; i < 8; ++i) sB[(grp * 8 + i) * 34 + row] = u.e[i];
    }
#if HAVE_TDM
    __builtin_amdgcn_s_wait_tensorcnt(0);  // no-op for waves that issued nothing
#endif
    __syncthreads();

    v16bf a0 = load_afrag(sA, 34, wm, 0);
    v16bf a1 = load_afrag(sA, 34, wm + 16, 0);
    v16bf b0 = load_bfrag(sB, 34, wn, 0);
    v16bf b1 = load_bfrag(sB, 34, wn + 16, 0);
    acc[0][0] = wmma_bf16(a0, b0, acc[0][0]);
    acc[0][1] = wmma_bf16(a0, b1, acc[0][1]);
    acc[1][0] = wmma_bf16(a1, b0, acc[1][0]);
    acc[1][1] = wmma_bf16(a1, b1, acc[1][1]);
  }

#pragma unroll
  for (int mi = 0; mi < 2; ++mi)
#pragma unroll
    for (int ni = 0; ni < 2; ++ni)
#pragma unroll
      for (int r = 0; r < 8; ++r) {
        int row = mBase + wm + mi * 16 + g * 8 + r;  // C/D layout: m = r + 8g
        int col = nBase + wn + ni * 16 + n16;
        float v = acc[mi][ni][r] + bias[col];
        if (EPI == 0) {
          int which = col >> 10;          // 0=Q,1=K,2=V
          int h = (col & 1023) >> 6;      // head
          int d = col & 63;               // within-head dim
          int b = row >> 11;              // T = 2048
          int t = row & 2047;
          size_t idx = (((size_t)(b * NH + h) * TT) + t) * DH + d;
          unsigned short bv = f32_to_bf16(v);
          if (which == 0)      outQ[idx] = bv;
          else if (which == 1) outK[idx] = bv;
          else                 outV[idx] = bv;
        } else {
          outF[(size_t)row * N + col] = v;
        }
      }
}

// ---------------------------------------------------------------------------
// Flash attention: one block per (b,h, 128 queries). Each wave owns 16 query
// rows (no cross-wave softmax reduction); block shares 32-key K/V LDS tiles.
// K tile (row-major, == B^T for Q*K^T) fetched by the TDM; V tile must land
// transposed, so the waves stage it while the TDM runs.
// ---------------------------------------------------------------------------
__global__ __launch_bounds__(256) void flash_attn_kernel(
    const unsigned short* __restrict__ Qb,
    const unsigned short* __restrict__ Kb,
    const unsigned short* __restrict__ Vb,
    unsigned short* __restrict__ Yb) {
  __shared__ unsigned short sK[32 * 66];        // keys x dh (TDM pad stride 66)
  __shared__ unsigned short sV[64 * 34];        // dh x keys (== B^T for P*V)
  __shared__ unsigned short sP[8 * 16 * 34];    // per-wave P scratch (C -> A relayout)

  const int tid  = threadIdx.x;
  const int lane = tid & 31, wid = tid >> 5;
  const int g = lane >> 4, n16 = lane & 15;
  const int bh = blockIdx.y;
  const int b = bh >> 4, h = bh & 15;
  const int qBase = blockIdx.x * 128;
  const size_t headBase = (size_t)bh * TT * DH;

  // Q fragments for this wave's 16 rows (kept in registers for whole kernel)
  v16bf aQ0 = load_afrag(Qb + headBase, DH, qBase + wid * 16, 0);
  v16bf aQ1 = load_afrag(Qb + headBase, DH, qBase + wid * 16, 32);

  v8f o0 = {}, o1 = {}, o2 = {}, o3 = {};
  float mrow[8], lrow[8];
#pragma unroll
  for (int r = 0; r < 8; ++r) { mrow[r] = -1e30f; lrow[r] = 0.0f; }

  unsigned short* sPw = sP + wid * (16 * 34);

  for (int kt = 0; kt < TT / 32; ++kt) {
    __syncthreads();
#if HAVE_TDM
    if (wid == 0) {
      tdm_load_2d(lds_off_of(sK), Kb + headBase + (size_t)kt * 32 * DH,
                  /*tensor_d0=*/DH, /*tensor_d1=*/32,
                  /*tile_d0=*/DH, /*tile_d1=*/32,
                  /*stride0=*/DH, TDM_FLAGS_ROW64);
    }
#endif
    {
      int row = tid >> 3, grp = tid & 7;
      size_t src = headBase + (size_t)(kt * 32 + row) * DH + grp * 8;
#if !HAVE_TDM
      uint4 dk = *(const uint4*)(Kb + src);
      unsigned int* dstk = (unsigned int*)(sK + row * 66 + grp * 8);
      dstk[0] = dk.x; dstk[1] = dk.y; dstk[2] = dk.z; dstk[3] = dk.w;
#endif
      union { uint4 q; unsigned short e[8]; } uv;
      uv.q = *(const uint4*)(Vb + src);
#pragma unroll
      for (int i = 0; i < 8; ++i) sV[(grp * 8 + i) * 34 + row] = uv.e[i];
    }
#if HAVE_TDM
    __builtin_amdgcn_s_wait_tensorcnt(0);
#endif
    __syncthreads();

    // S = Q @ K^T  (dh=64 contraction -> 2 WMMAs per 16-key tile)
    v8f s0 = {}, s1 = {};
    s0 = wmma_bf16(aQ0, load_bfrag(sK, 66, 0, 0), s0);
    s0 = wmma_bf16(aQ1, load_bfrag(sK, 66, 0, 32), s0);
    s1 = wmma_bf16(aQ0, load_bfrag(sK, 66, 16, 0), s1);
    s1 = wmma_bf16(aQ1, load_bfrag(sK, 66, 16, 32), s1);

    // online softmax (fp32); 16-lane reductions stay inside each half-wave,
    // matching the C-fragment row split m = r + 8g.
#pragma unroll
    for (int r = 0; r < 8; ++r) {
      float tmax = fmaxf(s0[r], s1[r]);
#pragma unroll
      for (int msk = 1; msk < 16; msk <<= 1)
        tmax = fmaxf(tmax, __shfl_xor(tmax, msk, 32));
      float mnew  = fmaxf(mrow[r], tmax);
      float scale = __expf(mrow[r] - mnew);
      float e0 = __expf(s0[r] - mnew);
      float e1 = __expf(s1[r] - mnew);
      float rs = e0 + e1;
#pragma unroll
      for (int msk = 1; msk < 16; msk <<= 1)
        rs += __shfl_xor(rs, msk, 32);
      lrow[r] = lrow[r] * scale + rs;
      mrow[r] = mnew;
      o0[r] *= scale; o1[r] *= scale; o2[r] *= scale; o3[r] *= scale;
      int m = g * 8 + r;
      sPw[m * 34 + n16]      = f32_to_bf16(e0);
      sPw[m * 34 + 16 + n16] = f32_to_bf16(e1);
    }
    asm volatile("s_wait_dscnt 0x0" ::: "memory");  // P stores -> P frag reload
    v16bf aP = load_afrag(sPw, 34, 0, 0);

    // O += P @ V
    o0 = wmma_bf16(aP, load_bfrag(sV, 34, 0,  0), o0);
    o1 = wmma_bf16(aP, load_bfrag(sV, 34, 16, 0), o1);
    o2 = wmma_bf16(aP, load_bfrag(sV, 34, 32, 0), o2);
    o3 = wmma_bf16(aP, load_bfrag(sV, 34, 48, 0), o3);
  }

  // y = O / (l * sqrt(dh)), stored (b,t,h*64+d) bf16 for the output GEMM
#pragma unroll
  for (int r = 0; r < 8; ++r) {
    float inv = 1.0f / (lrow[r] * 8.0f);
    int m = g * 8 + r;
    int t = qBase + wid * 16 + m;
    size_t rowOff = ((size_t)b * TT + t) * CC + h * DH;
    Yb[rowOff + 0  + n16] = f32_to_bf16(o0[r] * inv);
    Yb[rowOff + 16 + n16] = f32_to_bf16(o1[r] * inv);
    Yb[rowOff + 32 + n16] = f32_to_bf16(o2[r] * inv);
    Yb[rowOff + 48 + n16] = f32_to_bf16(o3[r] * inv);
  }
}

extern "C" void kernel_launch(void* const* d_in, const int* in_sizes, int n_in,
                              void* d_out, int out_size, void* d_ws, size_t ws_size,
                              hipStream_t stream) {
  (void)in_sizes; (void)n_in; (void)out_size; (void)ws_size;
  const float* x      = (const float*)d_in[0];
  const float* W_attn = (const float*)d_in[1];
  const float* b_attn = (const float*)d_in[2];
  const float* W_proj = (const float*)d_in[3];
  const float* b_proj = (const float*)d_in[4];

  // Workspace carve (bf16 halfwords), ~50 MB total
  unsigned short* Xbf = (unsigned short*)d_ws;               // 4096x1024
  unsigned short* Wab = Xbf + (size_t)4096 * 1024;           // 1024x3072
  unsigned short* Wpb = Wab + (size_t)1024 * 3072;           // 1024x1024
  unsigned short* Qb  = Wpb + (size_t)1024 * 1024;           // 32 heads x 2048 x 64
  unsigned short* Kb  = Qb  + (size_t)32 * TT * DH;
  unsigned short* Vb  = Kb  + (size_t)32 * TT * DH;
  unsigned short* Yb  = Vb  + (size_t)32 * TT * DH;          // 4096x1024

  cvt_f32_bf16_kernel<<<2048, 256, 0, stream>>>(x, Xbf, 4096 * 1024);
  cvt_f32_bf16_kernel<<<2048, 256, 0, stream>>>(W_attn, Wab, 1024 * 3072);
  cvt_f32_bf16_kernel<<<1024, 256, 0, stream>>>(W_proj, Wpb, 1024 * 1024);

  // QKV projection: [4096,1024] x [1024,3072] -> head-major Q/K/V (bf16)
  gemm_bf16_kernel<0><<<dim3(48, 32), 256, 0, stream>>>(
      Xbf, Wab, b_attn, 3072, 1024, Qb, Kb, Vb, nullptr);

  // Attention: grid = (q-tiles, b*h)
  flash_attn_kernel<<<dim3(TT / 128, 32), 256, 0, stream>>>(Qb, Kb, Vb, Yb);

  // Output projection: [4096,1024] x [1024,1024] + bias -> fp32 d_out
  gemm_bf16_kernel<1><<<dim3(16, 32), 256, 0, stream>>>(
      Yb, Wpb, b_proj, 1024, 1024, nullptr, nullptr, nullptr, (float*)d_out);
}
